// MoEGenreClassifier_39737037423283
// MI455X (gfx1250) — matrix operand order repaired and checked by reading, not compiled, not run on hardware
//
#include <hip/hip_runtime.h>
#include <hip/hip_bf16.h>

// ---------------------------------------------------------------------------
// MoE transformer forward for MI455X (gfx1250, wave32, WMMA).
// All matmuls -> v_wmma_f32_16x16x32_bf16 (bf16 in, fp32 acc).
// - 128x64x32 block tile, 8 waves, 2x2 16x16 tiles per wave (4 WMMA/k-step),
//   double-buffered LDS, one barrier per k-step.
// - 16B vectorized staging (global_load_b128), loop-invariant gather offsets.
// - MoE: top-1 atomic compaction, gather-A / scatter-C rows, device row count.
// ---------------------------------------------------------------------------

typedef long long i64;
typedef __bf16 v16bf __attribute__((ext_vector_type(16)));
typedef float  v8f   __attribute__((ext_vector_type(8)));
typedef __bf16 bf8v  __attribute__((ext_vector_type(8)));   // 16B
typedef float  f4v   __attribute__((ext_vector_type(4)));   // 16B

// Model dims
constexpr int cB = 32, cS = 512, cD = 768, cH = 12, cHD = 64;
constexpr int cF = 3072, cE = 8, cC = 10;
constexpr int cT = cB * cS;          // 16384 tokens
constexpr int cD3 = 3 * cD;          // 2304
constexpr int CHUNK = 4096;          // FFN m-chunk (bounds h1 workspace)
constexpr int cG = 4;                // attention batch group

#define BM 128
#define BN 64
#define BK 32

enum { ACT_NONE = 0, ACT_SILU = 1, ACT_RELU = 2 };

__device__ inline bf8v bf8_zero() {
  bf8v r;
  #pragma unroll
  for (int j = 0; j < 8; ++j) r[j] = (__bf16)0.0f;
  return r;
}
// 8-element weight-row segment load; fp32 converted to bf16 in-flight.
__device__ inline bf8v load_w8(const float* p) {
  const f4v* q = (const f4v*)p;
  f4v x = q[0], y = q[1];
  bf8v r;
  #pragma unroll
  for (int j = 0; j < 4; ++j) { r[j] = (__bf16)x[j]; r[4 + j] = (__bf16)y[j]; }
  return r;
}
__device__ inline bf8v load_w8(const __bf16* p) { return *(const bf8v*)p; }

// ---------------------------------------------------------------------------
// NT GEMM: C[m,n] = act( scale * sum_k A[m,k]*W[n,k] + bias[n] )
// Requirements (all call sites satisfy): K % 32 == 0, lda/ldw % 8 == 0,
// 16B-aligned base pointers. Batch: blockIdx.z -> (z0 = z % Z0, z1 = z / Z0)
// with independent element strides (used for [group][head] attention batching).
// ---------------------------------------------------------------------------
template <typename WT>
__global__ __launch_bounds__(256) void gemm_nt_kernel(
    const __bf16* __restrict__ A, i64 lda, i64 sAz0, i64 sAz1,
    const WT* __restrict__ W, i64 ldw, i64 sWz0, i64 sWz1,
    const float* __restrict__ bias,
    float* __restrict__ Cf, __bf16* __restrict__ Cb, i64 ldc, i64 sCz0, i64 sCz1,
    int M, int N, int K, int Z0, float scale, int act,
    const int* __restrict__ gatherA, const int* __restrict__ scatterC,
    const int* __restrict__ mcount, int mbase)
{
  constexpr int PK = BK + 8;   // 40 elems = 80B row: 16B-aligned, 20-bank stagger
  __shared__ __bf16 As[2][BM * PK];
  __shared__ __bf16 Bs[2][BN * PK];

  int rows = M;
  if (mcount) {                       // device-side row count (MoE capacity)
    int c = *mcount - mbase;
    rows = c < 0 ? 0 : (c > M ? M : c);
  }
  const int m0 = blockIdx.y * BM;
  const int n0 = blockIdx.x * BN;
  if (m0 >= rows) return;             // block-uniform early exit

  const int z0 = (int)(blockIdx.z % (unsigned)Z0);
  const int z1 = (int)(blockIdx.z / (unsigned)Z0);
  A += (i64)z0 * sAz0 + (i64)z1 * sAz1;
  W += (i64)z0 * sWz0 + (i64)z1 * sWz1;
  const i64 coff = (i64)z0 * sCz0 + (i64)z1 * sCz1;

  const int tid  = threadIdx.x;
  const int lane = tid & 31;
  const int wave = tid >> 5;
  const int wmB  = (wave & 3) * 32;   // base of wave's 2 row tiles
  const int wnB  = (wave >> 2) * 32;  // base of wave's 2 col tiles
  const int hl   = lane >> 4;         // lane half (0/1)
  const int l16  = lane & 15;

  // Per-thread staging: rows ar and ar+64 of A, row ar of W; 8 cols from ac.
  const int ar = tid >> 2;
  const int ac = (tid & 3) * 8;

  // Loop-invariant global offsets (gather lookups hoisted out of the k-loop).
  const bool aV0 = (m0 + ar) < rows;
  const bool aV1 = (m0 + ar + 64) < rows;
  i64 aOff0 = 0, aOff1 = 0;
  if (aV0) {
    i64 row = gatherA ? (i64)gatherA[mbase + m0 + ar] : (i64)(m0 + ar);
    aOff0 = row * lda + ac;
  }
  if (aV1) {
    i64 row = gatherA ? (i64)gatherA[mbase + m0 + ar + 64] : (i64)(m0 + ar + 64);
    aOff1 = row * lda + ac;
  }
  const bool wValid = (n0 + ar) < N;
  const i64 wOff = (i64)(n0 + ar) * ldw + ac;

  v8f acc[2][2];
  #pragma unroll
  for (int mt = 0; mt < 2; ++mt)
    #pragma unroll
    for (int nt = 0; nt < 2; ++nt)
      acc[mt][nt] = (v8f){0, 0, 0, 0, 0, 0, 0, 0};

  const int nk = K / BK;

  // prologue: tile 0 -> regs -> LDS buf 0
  bf8v a0 = aV0 ? *(const bf8v*)(A + aOff0) : bf8_zero();
  bf8v a1 = aV1 ? *(const bf8v*)(A + aOff1) : bf8_zero();
  bf8v wr = wValid ? load_w8(W + wOff) : bf8_zero();
  *(bf8v*)&As[0][ar * PK + ac] = a0;
  *(bf8v*)&As[0][(ar + 64) * PK + ac] = a1;
  *(bf8v*)&Bs[0][ar * PK + ac] = wr;
  __syncthreads();

  for (int ki = 0; ki < nk; ++ki) {
    const int cur = ki & 1;
    const bool more = (ki + 1) < nk;
    if (more) {                        // overlap next tile's loads with WMMA
      const int k1 = (ki + 1) * BK;
      a0 = aV0 ? *(const bf8v*)(A + aOff0 + k1) : bf8_zero();
      a1 = aV1 ? *(const bf8v*)(A + aOff1 + k1) : bf8_zero();
      wr = wValid ? load_w8(W + wOff + k1) : bf8_zero();
      if (wValid && (ki + 2) < nk)
        __builtin_prefetch(W + wOff + (i64)(ki + 2) * BK, 0, 1);
    }

    // WMMA fragments per documented wave32 layouts.
    // A 16x32: lanes 0-15 row M=l16, K={0..7,16..23}; lanes 16-31 K={8..15,24..31}.
    // B 32x16: lane col N=l16, K same order as A.
    v16bf af0, af1, bfr0, bfr1;
    #pragma unroll
    for (int j = 0; j < 16; ++j) {
      int ka = (j < 8) ? (hl * 8 + j) : (16 + hl * 8 + (j - 8));
      af0[j]  = As[cur][(wmB + l16) * PK + ka];
      af1[j]  = As[cur][(wmB + 16 + l16) * PK + ka];
      bfr0[j] = Bs[cur][(wnB + l16) * PK + ka];
      bfr1[j] = Bs[cur][(wnB + 16 + l16) * PK + ka];
    }
    acc[0][0] = __builtin_amdgcn_wmma_f32_16x16x32_bf16(false, af0, false, bfr0,
                                                        (short)0, acc[0][0], false, false);
    acc[0][1] = __builtin_amdgcn_wmma_f32_16x16x32_bf16(false, af0, false, bfr1,
                                                        (short)0, acc[0][1], false, false);
    acc[1][0] = __builtin_amdgcn_wmma_f32_16x16x32_bf16(false, af1, false, bfr0,
                                                        (short)0, acc[1][0], false, false);
    acc[1][1] = __builtin_amdgcn_wmma_f32_16x16x32_bf16(false, af1, false, bfr1,
                                                        (short)0, acc[1][1], false, false);

    if (more) {                        // stage next tile into the other buffer
      *(bf8v*)&As[cur ^ 1][ar * PK + ac] = a0;
      *(bf8v*)&As[cur ^ 1][(ar + 64) * PK + ac] = a1;
      *(bf8v*)&Bs[cur ^ 1][ar * PK + ac] = wr;
    }
    __syncthreads();
  }

  // Epilogue: C layout lanes 0-15 -> M=r, lanes 16-31 -> M=8+r.
  #pragma unroll
  for (int mt = 0; mt < 2; ++mt) {
    #pragma unroll
    for (int nt = 0; nt < 2; ++nt) {
      int gn = n0 + wnB + nt * 16 + l16;
      if (gn >= N) continue;
      float bv = bias ? bias[gn] : 0.0f;
      #pragma unroll
      for (int r = 0; r < 8; ++r) {
        int gm = m0 + wmB + mt * 16 + hl * 8 + r;
        if (gm >= rows) continue;
        float v = acc[mt][nt][r] * scale + bv;
        if (act == ACT_SILU)      v = v / (1.0f + __expf(-v));
        else if (act == ACT_RELU) v = fmaxf(v, 0.0f);
        i64 crow = scatterC ? (i64)scatterC[mbase + gm] : (i64)gm;
        i64 o = coff + crow * ldc + gn;
        if (Cf) Cf[o] = v;
        if (Cb) Cb[o] = (__bf16)v;
      }
    }
  }
}

// ---------------------------------------------------------------------------
// Embedding: x[t] = token_emb[ids[t]] + pos_emb[t % S]; fp32 + bf16 copies.
// ---------------------------------------------------------------------------
__global__ __launch_bounds__(256) void embed_kernel(
    const int* __restrict__ ids, const float* __restrict__ tok,
    const float* __restrict__ pos, float* __restrict__ xf,
    __bf16* __restrict__ xb)
{
  const i64 t = blockIdx.x;
  const int id = ids[t];
  const int s = (int)(t % cS);
  for (int d = threadIdx.x; d < cD; d += blockDim.x) {
    float v = tok[(i64)id * cD + d] + pos[(i64)s * cD + d];
    xf[t * cD + d] = v;
    xb[t * cD + d] = (__bf16)v;
  }
}

// ---------------------------------------------------------------------------
// x_out = LayerNorm(x + r) * g + b (fp32 stats, eps=1e-5); bf16 mirror.
// ---------------------------------------------------------------------------
__global__ __launch_bounds__(256) void resid_ln_kernel(
    const float* __restrict__ x, const float* __restrict__ r,
    const float* __restrict__ g, const float* __restrict__ be,
    float* __restrict__ xo, __bf16* __restrict__ xob)
{
  const i64 t = blockIdx.x;
  const int tid = threadIdx.x;
  __shared__ float red[256];
  float v[3];
  float s = 0.0f;
  #pragma unroll
  for (int i = 0; i < 3; ++i) {
    int d = tid + i * 256;
    float a = x[t * cD + d] + r[t * cD + d];
    v[i] = a; s += a;
  }
  red[tid] = s; __syncthreads();
  for (int o = 128; o > 0; o >>= 1) { if (tid < o) red[tid] += red[tid + o]; __syncthreads(); }
  const float mean = red[0] * (1.0f / cD);
  __syncthreads();
  float vs = 0.0f;
  #pragma unroll
  for (int i = 0; i < 3; ++i) { float d0 = v[i] - mean; vs += d0 * d0; }
  red[tid] = vs; __syncthreads();
  for (int o = 128; o > 0; o >>= 1) { if (tid < o) red[tid] += red[tid + o]; __syncthreads(); }
  const float rstd = rsqrtf(red[0] * (1.0f / cD) + 1e-5f);
  #pragma unroll
  for (int i = 0; i < 3; ++i) {
    int d = tid + i * 256;
    float o2 = (v[i] - mean) * rstd * g[d] + be[d];
    xo[t * cD + d] = o2;
    xob[t * cD + d] = (__bf16)o2;
  }
}

// ---------------------------------------------------------------------------
// Row softmax over length 512; fp32 in, bf16 out.
// ---------------------------------------------------------------------------
__global__ __launch_bounds__(256) void softmax_kernel(
    const float* __restrict__ sc, __bf16* __restrict__ at)
{
  const i64 row = blockIdx.x;
  const float* p = sc + row * cS;
  __bf16* o = at + row * cS;
  const int tid = threadIdx.x;
  __shared__ float red[256];
  float a = p[tid], b = p[tid + 256];
  red[tid] = fmaxf(a, b); __syncthreads();
  for (int off = 128; off > 0; off >>= 1) { if (tid < off) red[tid] = fmaxf(red[tid], red[tid + off]); __syncthreads(); }
  const float mx = red[0];
  __syncthreads();
  float ea = __expf(a - mx), eb = __expf(b - mx);
  red[tid] = ea + eb; __syncthreads();
  for (int off = 128; off > 0; off >>= 1) { if (tid < off) red[tid] += red[tid + off]; __syncthreads(); }
  const float inv = 1.0f / red[0];
  o[tid] = (__bf16)(ea * inv);
  o[tid + 256] = (__bf16)(eb * inv);
}

// ---------------------------------------------------------------------------
// V transpose for a group of cG batches:
// vt[(g*H*HD + h*HD + d)*S + s] = qkv_g[g*S*D3 + s*D3 + 2D + h*HD + d]
// ---------------------------------------------------------------------------
__global__ __launch_bounds__(256) void vtrans_kernel(
    const __bf16* __restrict__ qkv_g, __bf16* __restrict__ vt)
{
  const int gx = blockIdx.x;                 // 0 .. cG*H*HD-1
  const int g  = gx / (cH * cHD);
  const int hd = gx % (cH * cHD);
  const __bf16* src = qkv_g + (i64)g * cS * cD3 + 2 * cD + hd;
  __bf16* dst = vt + (i64)gx * cS;
  for (int s = threadIdx.x; s < cS; s += blockDim.x)
    dst[s] = src[(i64)s * cD3];
}

// ---------------------------------------------------------------------------
// Top-1 router: softmax over 8 logits, first-argmax, atomic compaction,
// importance/load accumulation.
// ---------------------------------------------------------------------------
__global__ __launch_bounds__(256) void route_kernel(
    const float* __restrict__ logits, int* __restrict__ lists,
    int* __restrict__ counts, float* __restrict__ imp)
{
  __shared__ float pimp[cE];
  const int tid = threadIdx.x;
  if (tid < cE) pimp[tid] = 0.0f;
  __syncthreads();
  const int t = blockIdx.x * 256 + tid;
  if (t < cT) {
    float l[cE];
    float mx = -1e30f; int am = 0;
    #pragma unroll
    for (int e = 0; e < cE; ++e) {
      l[e] = logits[(i64)t * cE + e];
      if (l[e] > mx) { mx = l[e]; am = e; }   // strict > keeps first max
    }
    float s = 0.0f;
    #pragma unroll
    for (int e = 0; e < cE; ++e) { l[e] = __expf(l[e] - mx); s += l[e]; }
    const float inv = 1.0f / s;
    #pragma unroll
    for (int e = 0; e < cE; ++e) atomicAdd(&pimp[e], l[e] * inv);
    int pos = atomicAdd(&counts[am], 1);
    lists[am * cT + pos] = t;
  }
  __syncthreads();
  if (tid < cE) atomicAdd(&imp[tid], pimp[tid]);
}

__global__ void aux_kernel(const float* __restrict__ imp,
                           const int* __restrict__ counts,
                           float* __restrict__ out_aux)
{
  if (threadIdx.x == 0 && blockIdx.x == 0) {
    const float tinv = 1.0f / (float)cT;
    float s = 0.0f;
    for (int e = 0; e < cE; ++e)
      s += (imp[e] * tinv) * ((float)counts[e] * tinv);
    atomicAdd(out_aux, (float)cE * s);
  }
}

// ---------------------------------------------------------------------------
// Masked mean pooling: rep[b] = sum_s x[b,s]*m / max(sum m, 1e-6)
// ---------------------------------------------------------------------------
__global__ __launch_bounds__(256) void pool_kernel(
    const float* __restrict__ x, const int* __restrict__ mask,
    float* __restrict__ rep, __bf16* __restrict__ repb)
{
  const int b = blockIdx.x;
  const int tid = threadIdx.x;
  __shared__ float red[256];
  float ms = 0.0f;
  for (int s = tid; s < cS; s += 256) ms += (float)mask[b * cS + s];
  red[tid] = ms; __syncthreads();
  for (int o = 128; o > 0; o >>= 1) { if (tid < o) red[tid] += red[tid + o]; __syncthreads(); }
  const float denom = fmaxf(red[0], 1e-6f);
  __syncthreads();
  for (int d = tid; d < cD; d += 256) {
    float s = 0.0f;
    for (int t2 = 0; t2 < cS; ++t2) {
      float m = (float)mask[b * cS + t2];
      s += x[((i64)b * cS + t2) * cD + d] * m;
    }
    float v = s / denom;
    rep[(i64)b * cD + d] = v;
    repb[(i64)b * cD + d] = (__bf16)v;
  }
}

__global__ void zero_small_kernel(float* f, int nf, int* ip, int ni)
{
  const int t = threadIdx.x;
  if (f && t < nf) f[t] = 0.0f;
  if (ip && t < ni) ip[t] = 0;
}

// ---------------------------------------------------------------------------
// Host side
// ---------------------------------------------------------------------------
template <typename WT>
static void launch_gemm(hipStream_t st,
                        const __bf16* A, i64 lda, i64 sAz0, i64 sAz1,
                        const WT* W, i64 ldw, i64 sWz0, i64 sWz1,
                        const float* bias,
                        float* Cf, __bf16* Cb, i64 ldc, i64 sCz0, i64 sCz1,
                        int M, int N, int K, int Z0, int Z, float scale, int act,
                        const int* gatherA = nullptr, const int* scatterC = nullptr,
                        const int* mcount = nullptr, int mbase = 0)
{
  dim3 g((N + BN - 1) / BN, (M + BM - 1) / BM, Z);
  gemm_nt_kernel<WT><<<g, 256, 0, st>>>(A, lda, sAz0, sAz1, W, ldw, sWz0, sWz1,
                                        bias, Cf, Cb, ldc, sCz0, sCz1,
                                        M, N, K, Z0, scale, act,
                                        gatherA, scatterC, mcount, mbase);
}

extern "C" void kernel_launch(void* const* d_in, const int* in_sizes, int n_in,
                              void* d_out, int out_size, void* d_ws, size_t ws_size,
                              hipStream_t stream)
{
  (void)in_sizes; (void)n_in; (void)out_size; (void)ws_size;

  // Input flattening: JAX pytree order (dicts sorted by key).
  //  0 input_ids, 1 attention_mask, 2 c1_b, 3 c1_w, 4 c2_b, 5 c2_w,
  //  6.. layers, 84 pos_emb, 85 token_emb
  auto F32 = [&](int i) { return (const float*)d_in[i]; };
  auto I32 = [&](int i) { return (const int*)d_in[i]; };

  const int* ids  = I32(0);
  const int* mask = I32(1);
  const float* c1_b = F32(2); const float* c1_w = F32(3);
  const float* c2_b = F32(4); const float* c2_w = F32(5);
  const int  lbase[6]  = {6, 18, 32, 44, 58, 70};
  const bool is_moe[6] = {false, true, false, true, false, true};
  const float* pos_emb = F32(84);
  const float* tok_emb = F32(85);
  float* out = (float*)d_out;                 // [B*C logits][aux]

  // ---- workspace bump allocator ----
  char* ws = (char*)d_ws;
  size_t woff = 0;
  auto walloc = [&](size_t bytes) -> void* {
    void* p = ws + woff;
    woff = (woff + bytes + 255) & ~(size_t)255;
    return p;
  };
  float*  x_f32   = (float*) walloc((size_t)cT * cD * 4);
  __bf16* x_bf    = (__bf16*)walloc((size_t)cT * cD * 2);
  __bf16* qkv_bf  = (__bf16*)walloc((size_t)cT * cD3 * 2);
  __bf16* o_bf    = (__bf16*)walloc((size_t)cT * cD * 2);
  float*  res_f32 = (float*) walloc((size_t)cT * cD * 4);
  float*  scores  = (float*) walloc((size_t)cG * cH * cS * cS * 4);
  __bf16* attn_bf = (__bf16*)walloc((size_t)cG * cH * cS * cS * 2);
  __bf16* vt_bf   = (__bf16*)walloc((size_t)cG * cH * cHD * cS * 2);
  __bf16* h1_bf   = (__bf16*)walloc((size_t)CHUNK * cF * 2);
  float*  glog    = (float*) walloc((size_t)cT * cE * 4);
  int*    lists   = (int*)   walloc((size_t)cE * cT * 4);
  int*    counts  = (int*)   walloc((size_t)cE * 4);
  float*  imp     = (float*) walloc((size_t)cE * 4);
  float*  rep     = (float*) walloc((size_t)cB * cD * 4);
  __bf16* rep_bf  = (__bf16*)walloc((size_t)cB * cD * 2);
  __bf16* hrep_bf = (__bf16*)walloc((size_t)cB * cD * 2);

  // zero the aux accumulator slot
  zero_small_kernel<<<1, 32, 0, stream>>>(out + cB * cC, 1, nullptr, 0);

  // ---- embeddings ----
  embed_kernel<<<cT, 256, 0, stream>>>(ids, tok_emb, pos_emb, x_f32, x_bf);

  for (int L = 0; L < 6; ++L) {
    const int lb = lbase[L];
    const float *in_b = F32(lb + 0), *in_w = F32(lb + 1);
    const float *out_b = F32(lb + 2), *out_w = F32(lb + 3);
    const float *b1 = nullptr, *b2 = nullptr, *w1 = nullptr, *w2 = nullptr;
    const float *gate_b = nullptr, *gate_w = nullptr;
    const float *ln1_b, *ln1_g, *ln2_b, *ln2_g;
    if (!is_moe[L]) {
      b1 = F32(lb + 4); b2 = F32(lb + 5); w1 = F32(lb + 6); w2 = F32(lb + 7);
      ln1_b = F32(lb + 8); ln1_g = F32(lb + 9); ln2_b = F32(lb + 10); ln2_g = F32(lb + 11);
    } else {
      b1 = F32(lb + 4); b2 = F32(lb + 5); gate_b = F32(lb + 6); gate_w = F32(lb + 7);
      w1 = F32(lb + 8); w2 = F32(lb + 9);
      ln1_b = F32(lb + 10); ln1_g = F32(lb + 11); ln2_b = F32(lb + 12); ln2_g = F32(lb + 13);
    }

    // ---- fused QKV projection: [T,3D] bf16 ----
    launch_gemm<float>(stream, x_bf, cD, 0, 0, in_w, cD, 0, 0, in_b,
                       nullptr, qkv_bf, cD3, 0, 0,
                       cT, cD3, cD, 1, 1, 1.0f, ACT_NONE);

    // ---- attention, groups of cG batches; grid.z = cG*H (z0=head, z1=group) --
    for (int bg = 0; bg < cB; bg += cG) {
      const __bf16* qkv_g = qkv_bf + (i64)bg * cS * cD3;
      vtrans_kernel<<<cG * cH * cHD, 256, 0, stream>>>(qkv_g, vt_bf);
      // scores[g,h] = (Q · K^T) / 8
      launch_gemm<__bf16>(stream,
                          qkv_g, cD3, cHD, (i64)cS * cD3,
                          qkv_g + cD, cD3, cHD, (i64)cS * cD3, nullptr,
                          scores, nullptr, cS, (i64)cS * cS, (i64)cH * cS * cS,
                          cS, cS, cHD, cH, cG * cH, 0.125f, ACT_NONE);
      softmax_kernel<<<cG * cH * cS, 256, 0, stream>>>(scores, attn_bf);
      // O[g,h] = P · V  (W = V^T [HD,S])
      launch_gemm<__bf16>(stream,
                          attn_bf, cS, (i64)cS * cS, (i64)cH * cS * cS,
                          vt_bf, cS, (i64)cHD * cS, (i64)cH * cHD * cS, nullptr,
                          nullptr, o_bf + (i64)bg * cS * cD, cD, cHD, (i64)cS * cD,
                          cS, cHD, cS, cH, cG * cH, 1.0f, ACT_NONE);
    }

    // ---- output projection + residual LN1 ----
    launch_gemm<float>(stream, o_bf, cD, 0, 0, out_w, cD, 0, 0, out_b,
                       res_f32, nullptr, cD, 0, 0,
                       cT, cD, cD, 1, 1, 1.0f, ACT_NONE);
    resid_ln_kernel<<<cT, 256, 0, stream>>>(x_f32, res_f32, ln1_g, ln1_b, x_f32, x_bf);

    // ---- FFN ----
    if (!is_moe[L]) {
      for (int ch = 0; ch < cT / CHUNK; ++ch) {
        const i64 m0 = (i64)ch * CHUNK;
        launch_gemm<float>(stream, x_bf + m0 * cD, cD, 0, 0, w1, cD, 0, 0, b1,
                           nullptr, h1_bf, cF, 0, 0,
                           CHUNK, cF, cD, 1, 1, 1.0f, ACT_SILU);
        launch_gemm<float>(stream, h1_bf, cF, 0, 0, w2, cF, 0, 0, b2,
                           res_f32 + m0 * cD, nullptr, cD, 0, 0,
                           CHUNK, cD, cF, 1, 1, 1.0f, ACT_NONE);
      }
    } else {
      zero_small_kernel<<<1, 32, 0, stream>>>(imp, cE, counts, cE);
      launch_gemm<float>(stream, x_bf, cD, 0, 0, gate_w, cD, 0, 0, gate_b,
                         glog, nullptr, cE, 0, 0,
                         cT, cE, cD, 1, 1, 1.0f, ACT_NONE);
      route_kernel<<<cT / 256, 256, 0, stream>>>(glog, lists, counts, imp);
      aux_kernel<<<1, 32, 0, stream>>>(imp, counts, out + cB * cC);
      // per-expert compacted FFN; worst-case grids, device-count early exit
      for (int e = 0; e < cE; ++e) {
        const float* w1e = w1 + (i64)e * cF * cD;
        const float* w2e = w2 + (i64)e * cD * cF;
        const float* b1e = b1 + (i64)e * cF;
        const float* b2e = b2 + (i64)e * cD;
        const int* liste = lists + (i64)e * cT;
        const int* cnte  = counts + e;
        for (int ch = 0; ch < cT / CHUNK; ++ch) {
          const int mb = ch * CHUNK;
          launch_gemm<float>(stream, x_bf, cD, 0, 0, w1e, cD, 0, 0, b1e,
                             nullptr, h1_bf, cF, 0, 0,
                             CHUNK, cF, cD, 1, 1, 1.0f, ACT_SILU,
                             liste, nullptr, cnte, mb);
          launch_gemm<float>(stream, h1_bf, cF, 0, 0, w2e, cF, 0, 0, b2e,
                             res_f32, nullptr, cD, 0, 0,
                             CHUNK, cD, cF, 1, 1, 1.0f, ACT_NONE,
                             nullptr, liste, cnte, mb);
        }
      }
    }
    resid_ln_kernel<<<cT, 256, 0, stream>>>(x_f32, res_f32, ln2_g, ln2_b, x_f32, x_bf);
  }

  // ---- pooled classifier head ----
  pool_kernel<<<cB, 256, 0, stream>>>(x_f32, mask, rep, rep_bf);
  launch_gemm<float>(stream, rep_bf, cD, 0, 0, c1_w, cD, 0, 0, c1_b,
                     nullptr, hrep_bf, cD, 0, 0,
                     cB, cD, cD, 1, 1, 1.0f, ACT_RELU);
  launch_gemm<float>(stream, hrep_bf, cD, 0, 0, c2_w, cD, 0, 0, c2_b,
                     out, nullptr, cC, 0, 0,
                     cB, cC, cD, 1, 1, 1.0f, ACT_NONE);
}
